// AttentionLayer_22119081574472
// MI455X (gfx1250) — matrix operand order repaired and checked by reading, not compiled, not run on hardware
//
#include <hip/hip_runtime.h>
#include <math.h>

// ---------------------------------------------------------------------------
// Problem constants (reference): B=16, TQ=256, TK=256, QD=512, KD=512
// ---------------------------------------------------------------------------
#define BB    16
#define TQ    256
#define TK    256
#define QD    512
#define KD    512
#define MROWS (BB * TQ)            // 4096 flattened (b, t_q) rows

typedef __attribute__((ext_vector_type(16))) _Float16 v16h;
typedef __attribute__((ext_vector_type(8)))  _Float16 v8h;
typedef __attribute__((ext_vector_type(4)))  _Float16 v4h;
typedef __attribute__((ext_vector_type(8)))  float    v8f;
typedef __attribute__((ext_vector_type(4)))  int      v4i;

#define AS1 __attribute__((address_space(1)))
#define AS3 __attribute__((address_space(3)))

// gfx1250 native V_TANH_F32 when exposed; ocml fallback otherwise.
#if __has_builtin(__builtin_amdgcn_tanhf)
__device__ __forceinline__ float fast_tanh(float x) { return __builtin_amdgcn_tanhf(x); }
#else
__device__ __forceinline__ float fast_tanh(float x) { return tanhf(x); }
#endif

// gfx1250 async global->LDS copy (bypasses VGPRs, tracked by ASYNCcnt).
#if defined(__gfx1250__) && __has_builtin(__builtin_amdgcn_global_load_async_to_lds_b128) && __has_builtin(__builtin_amdgcn_s_wait_asynccnt)
#define HAVE_ASYNC_LDS 1
#else
#define HAVE_ASYNC_LDS 0
#endif

// ---------------------------------------------------------------------------
// Operand pre-conversion kernels (run once; ~11 MB total, trivial vs 23 TB/s)
// ---------------------------------------------------------------------------
__global__ __launch_bounds__(256) void cvt_f16(const float* __restrict__ in,
                                               _Float16* __restrict__ out, int n4) {
    int i = blockIdx.x * 256 + threadIdx.x;
    if (i >= n4) return;
    float4 v = ((const float4*)in)[i];
    v4h h;
    h[0] = (_Float16)v.x; h[1] = (_Float16)v.y;
    h[2] = (_Float16)v.z; h[3] = (_Float16)v.w;
    ((v4h*)out)[i] = h;
}

// out[c][r] = (f16) in[r][c]   (R x C input); blockIdx.y = batch
__global__ __launch_bounds__(256) void cvt_transpose_f16(
    const float* __restrict__ in, _Float16* __restrict__ out,
    int R, int C, long inB, long outB) {
    in  += (size_t)blockIdx.y * inB;
    out += (size_t)blockIdx.y * outB;
    int idx = blockIdx.x * 256 + threadIdx.x;
    if (idx >= R * C) return;
    int r = idx / C, c = idx - r * C;
    out[(size_t)c * R + r] = (_Float16)in[(size_t)r * C + c];
}

// ---------------------------------------------------------------------------
// Generic WMMA GEMM:  C[M,N] = act( A @ W^T' + bias )   (all operands f16)
//   * A  : f16 row-major, row stride == splitK; K in [0,splitK) reads A0,
//          [splitK,K) reads A1 (handles the [query; context] concat).
//   * WT : f16, pre-transposed to [N][K] so the B panel (64n x 32k = 4 KB)
//          stages into LDS via global_load_async_to_lds_b128 (ASYNCcnt).
//   * Block = 8 waves = 128 rows x 64 cols; wave = 16 x 64 (4 acc tiles
//     share one A fragment; A fragment = 2x global_load_b128, no cvt).
//   * Fragment layouts used here (16x16x32 f16; contiguous k order, same in A and B):
//       A: lane = half*16+m; elements = k run [half*16,+16)
//       B: element e -> k = half*16+e, n = lane&15  (contiguous in Bs[n][k])
//       C: m = half*8+r, n = lane&15
//   * ACT / OUTF16 are compile-time so the epilogue is straight-line stores.
// ---------------------------------------------------------------------------
template <bool ACT, bool OUTF16>
__global__ __launch_bounds__(256) void gemm_wmma(
    const _Float16* __restrict__ A0, const _Float16* __restrict__ A1, int splitK,
    const _Float16* __restrict__ WT, const float* __restrict__ bias,
    float* __restrict__ Cf, _Float16* __restrict__ Ch,
    int M, int N, int K, long aB, long wB, long cB)
{
    __shared__ _Float16 Bs[64 * 32];            // [n][k], 4 KB

    const int b = blockIdx.y;
    A0 += (size_t)b * aB;
    if (A1) A1 += (size_t)b * aB;
    WT += (size_t)b * wB;
    if (OUTF16) Ch += (size_t)b * cB;
    else        Cf += (size_t)b * cB;

    const int nblks = N / 64;
    const int bx   = blockIdx.x % nblks;
    const int by   = blockIdx.x / nblks;
    const int wave = threadIdx.x >> 5;
    const int lane = threadIdx.x & 31;
    const int half = lane >> 4;
    const int l16  = lane & 15;
    const int m0   = by * 128 + wave * 16;
    const int n0   = bx * 64;

    // bias column values, hoisted out of the loop (uniform over r)
    float bcol[4];
#pragma unroll
    for (int t = 0; t < 4; ++t)
        bcol[t] = bias ? bias[n0 + t * 16 + l16] : 0.f;

    // B staging: thread t copies 8 halves (16B): row n = t>>2, chunk (t&3)*8
    const int sn = threadIdx.x >> 2;
    const int sk = (threadIdx.x & 3) * 8;

    v8f acc[4] = {v8f{}, v8f{}, v8f{}, v8f{}};

    for (int k0 = 0; k0 < K; k0 += 32) {
        // ---- stage WT[n0+sn][k0+sk .. +8) -> Bs[sn][sk] (contiguous copy)
        const _Float16* bsrc = WT + (size_t)(n0 + sn) * K + k0 + sk;
        _Float16*       bdst = &Bs[sn * 32 + sk];
        __syncthreads();                         // prev-iter Bs readers done
#if HAVE_ASYNC_LDS
        __builtin_amdgcn_global_load_async_to_lds_b128(
            (AS1 v4i*)const_cast<_Float16*>(bsrc),
            (AS3 v4i*)bdst, 0, 0);
        __builtin_amdgcn_s_wait_asynccnt(0);
#else
        *(float4*)bdst = *(const float4*)bsrc;
#endif
        __syncthreads();

        // ---- A fragment: two contiguous 8-half runs -> 2x global_load_b128
        const _Float16* arow;
        int kb;
        if (k0 < splitK) { arow = A0; kb = k0; }
        else             { arow = A1; kb = k0 - splitK; }
        const _Float16* ap = arow + (size_t)(m0 + l16) * (size_t)splitK + kb;
        v8h alo = *(const v8h*)(ap + half * 16);
        v8h ahi = *(const v8h*)(ap + half * 16 + 8);
        v16h a;
#pragma unroll
        for (int e = 0; e < 8; ++e) { a[e] = alo[e]; a[8 + e] = ahi[e]; }

        // ---- prefetch all 4 B fragments, then 4 back-to-back WMMAs
        v16h bf[4];
#pragma unroll
        for (int t = 0; t < 4; ++t)
            bf[t] = *(const v16h*)&Bs[(t * 16 + l16) * 32 + half * 16];
#pragma unroll
        for (int t = 0; t < 4; ++t)
            acc[t] = __builtin_amdgcn_wmma_f32_16x16x32_f16(
                false, a, false, bf[t], (short)0, acc[t], false, false);
    }

    // ---- epilogue: straight-line, coalesced stores
#pragma unroll
    for (int r = 0; r < 8; ++r) {
        const int m = m0 + half * 8 + r;
#pragma unroll
        for (int t = 0; t < 4; ++t) {
            const int n = n0 + t * 16 + l16;
            float v = acc[t][r] + bcol[t];
            if (ACT) v = fast_tanh(v);
            if (OUTF16) Ch[(size_t)m * N + n] = (_Float16)v;
            else        Cf[(size_t)m * N + n] = v;
        }
    }
}

// ---------------------------------------------------------------------------
// Fused Bahdanau score + softmax (the 536M-tanh roofline stage; f32 math).
//   block = one (b,q) row; thread t owns k=t.
//   score[k] = sum_n tanh(aq[q,n] + keys[b,k,n]) * w_att[n] + b_att
// float4 loads, v_tanh_f32 + fma inner loop; aq/w_att in LDS; keys rows
// stream from L2 (512 KB/batch resident across the 256 q-rows reusing them).
// Emits p in f32 (output slice) and f16 (operand for the context BMM).
// ---------------------------------------------------------------------------
__global__ __launch_bounds__(256) void k2_score_softmax(
    const float* __restrict__ aq, const float* __restrict__ keys,
    const float* __restrict__ w_att, const float* __restrict__ b_att,
    float* __restrict__ p_out, _Float16* __restrict__ p_h)
{
    __shared__ __align__(16) float s_aq[KD];
    __shared__ __align__(16) float s_w[KD];
    __shared__ float red[TK];

    const int bq = blockIdx.x;                  // 0..4095
    const int b  = bq >> 8;
    const int t  = threadIdx.x;                 // k index, 0..255
    s_aq[t]       = aq[(size_t)bq * KD + t];
    s_aq[t + 256] = aq[(size_t)bq * KD + t + 256];
    s_w[t]        = w_att[t];
    s_w[t + 256]  = w_att[t + 256];
    __syncthreads();

    const float4* krow = (const float4*)(keys + (size_t)(b * TK + t) * KD);
    const float4* a4   = (const float4*)s_aq;
    const float4* w4   = (const float4*)s_w;
    float s = 0.f;
#pragma unroll 4
    for (int n = 0; n < KD / 4; ++n) {
        float4 kv = krow[n];
        float4 av = a4[n];
        float4 wv = w4[n];
        s = fmaf(fast_tanh(av.x + kv.x), wv.x, s);
        s = fmaf(fast_tanh(av.y + kv.y), wv.y, s);
        s = fmaf(fast_tanh(av.z + kv.z), wv.z, s);
        s = fmaf(fast_tanh(av.w + kv.w), wv.w, s);
    }
    s += b_att[0];

    // block softmax over 256 lanes
    red[t] = s;
    __syncthreads();
    for (int off = 128; off > 0; off >>= 1) {
        if (t < off) red[t] = fmaxf(red[t], red[t + off]);
        __syncthreads();
    }
    float mx = red[0];
    __syncthreads();
    float e = __expf(s - mx);
    red[t] = e;
    __syncthreads();
    for (int off = 128; off > 0; off >>= 1) {
        if (t < off) red[t] += red[t + off];
        __syncthreads();
    }
    float p = e * (1.f / red[0]);
    p_out[(size_t)bq * TK + t] = p;
    p_h[(size_t)bq * TK + t]   = (_Float16)p;
}

// ---------------------------------------------------------------------------
extern "C" void kernel_launch(void* const* d_in, const int* in_sizes, int n_in,
                              void* d_out, int out_size, void* d_ws, size_t ws_size,
                              hipStream_t stream) {
    const float* query = (const float*)d_in[0];
    const float* keys  = (const float*)d_in[1];
    const float* Wq    = (const float*)d_in[2];
    const float* bq    = (const float*)d_in[3];
    const float* w_att = (const float*)d_in[4];
    const float* b_att = (const float*)d_in[5];
    const float* Wout  = (const float*)d_in[6];
    const float* bout  = (const float*)d_in[7];

    float* ctx_out = (float*)d_out;                     // [B,TQ,QD] = 2,097,152
    float* p_out   = ctx_out + (size_t)BB * TQ * QD;    // [B,TQ,TK] = 1,048,576

    // workspace layout (bytes)
    char* ws = (char*)d_ws;
    float*    aq_ws   = (float*)ws;    ws += (size_t)MROWS * KD * 4;        // 8 MB
    _Float16* query_h = (_Float16*)ws; ws += (size_t)MROWS * QD * 2;        // 4 MB
    _Float16* ctx_h   = (_Float16*)ws; ws += (size_t)MROWS * KD * 2;        // 4 MB
    _Float16* p_h     = (_Float16*)ws; ws += (size_t)BB * TQ * TK * 2;      // 2 MB
    _Float16* keysT_h = (_Float16*)ws; ws += (size_t)BB * KD * TK * 2;      // 4 MB
    _Float16* WqT_h   = (_Float16*)ws; ws += (size_t)KD * QD * 2;           // 0.5 MB
    _Float16* WoutT_h = (_Float16*)ws; ws += (size_t)QD * (QD + KD) * 2;    // 1 MB

    const int threads = 256;                            // 8 waves (wave32)

    // 0) one-time operand conversions to f16 (W's and keys also transposed)
    cvt_f16<<<(MROWS * QD / 4 + 255) / 256, threads, 0, stream>>>(
        query, query_h, MROWS * QD / 4);
    cvt_transpose_f16<<<dim3((QD * KD + 255) / 256, 1), threads, 0, stream>>>(
        Wq, WqT_h, QD, KD, 0, 0);
    cvt_transpose_f16<<<dim3(((QD + KD) * QD + 255) / 256, 1), threads, 0, stream>>>(
        Wout, WoutT_h, QD + KD, QD, 0, 0);
    cvt_transpose_f16<<<dim3((TK * KD + 255) / 256, BB), threads, 0, stream>>>(
        keys, keysT_h, TK, KD, (long)TK * KD, (long)KD * TK);

    // 1) att_query = query @ Wq + bq : M=4096,N=512,K=512  (f32 out for k2)
    gemm_wmma<false, false><<<dim3((MROWS / 128) * (KD / 64), 1), threads, 0, stream>>>(
        query_h, nullptr, QD, WqT_h, bq, aq_ws, nullptr, MROWS, KD, QD, 0, 0, 0);

    // 2) fused score + softmax -> p (f32 output slice + f16 operand)
    k2_score_softmax<<<BB * TQ, threads, 0, stream>>>(
        aq_ws, keys, w_att, b_att, p_out, p_h);

    // 3) context = p @ keys (batched over 16): M=256,N=512,K=256 (f16 out)
    gemm_wmma<false, true><<<dim3((TQ / 128) * (KD / 64), BB), threads, 0, stream>>>(
        p_h, nullptr, TK, keysT_h, nullptr, nullptr, ctx_h, TQ, KD, TK,
        (long)TQ * TK, (long)KD * TK, (long)TQ * KD);

    // 4) out = tanh([query; context] @ Wout + bout): M=4096,N=512,K=1024
    gemm_wmma<true, false><<<dim3((MROWS / 128) * (QD / 64), 1), threads, 0, stream>>>(
        query_h, ctx_h, QD, WoutT_h, bout, ctx_out, nullptr, MROWS, QD, QD + KD,
        0, 0, 0);
}
